// MultiHeadAttention_68719476871
// MI455X (gfx1250) — matrix operand order repaired and checked by reading, not compiled, not run on hardware
//
#include <hip/hip_runtime.h>
#include <hip/hip_bf16.h>
#include <stdint.h>

// MHA: B=2, S=2048, H=16, DH=64, D=1024
constexpr int kH  = 16;
constexpr int kDH = 64;
constexpr int kD  = 1024;
constexpr int kB  = 2;
constexpr int kS  = 2048;
constexpr int kBS = kB * kS; // 4096

typedef __bf16 v16bf __attribute__((ext_vector_type(16)));
typedef float  v8f   __attribute__((ext_vector_type(8)));
typedef unsigned int u32x4 __attribute__((ext_vector_type(4)));
typedef float        f32x4 __attribute__((ext_vector_type(4)));

union Frag {
    v16bf v;
    u32x4 q[2];
};

__device__ __forceinline__ __bf16 f2bf(float f) {
    union { float f; uint32_t u; } c; c.f = f;
    uint32_t u = c.u + 0x7FFFu + ((c.u >> 16) & 1u); // round-to-nearest-even
    union { uint16_t s; __bf16 b; } r; r.s = (uint16_t)(u >> 16);
    return r.b;
}

__device__ __forceinline__ v8f wmma_bf16(v16bf a, v16bf b, v8f c) {
    // D(f32 16x16) = A(16x32 bf16) * B(32x16 bf16) + C
    return __builtin_amdgcn_wmma_f32_16x16x32_bf16(
        false, a, false, b, (short)0, c, false, false);
}

// A-matrix 16x32 bf16 fragment (ISA 7.12.2): lanes 0-15 row=lane, K {0..7,16..23};
// lanes 16-31 row=lane-15? -> row=lane&15, K {8..15,24..31}. Two 16B loads.
__device__ __forceinline__ v16bf load_afrag(const __bf16* base, int ldk,
                                            int row0, int k0, int lane) {
    int half = (lane & 16) ? 8 : 0;
    const __bf16* p = base + (size_t)(row0 + (lane & 15)) * ldk + k0 + half;
    Frag f;
    f.q[0] = *reinterpret_cast<const u32x4*>(p);
    f.q[1] = *reinterpret_cast<const u32x4*>(p + 16);
    return f.v;
}

// B-matrix 32x16 bf16 fragment: lanes 0-15 col=lane, K {0..7,16..23};
// lanes 16-31 col=lane&15, K {8..15,24..31}. baseT is B^T row-major [N][ldk]. Two 16B loads.
__device__ __forceinline__ v16bf load_bfrag(const __bf16* baseT, int ldk,
                                            int col0, int k0, int lane) {
    int half = (lane & 16) ? 8 : 0;
    const __bf16* p = baseT + (size_t)(col0 + (lane & 15)) * ldk + k0 + half;
    const u32x4* q = reinterpret_cast<const u32x4*>(p);
    Frag f;
    f.q[0] = q[0];
    f.q[1] = q[2];
    return f.v;
}

// ---------------- conversion kernels ----------------

__global__ __launch_bounds__(256) void cvt_x_kernel(const float* __restrict__ x,
                                                    __bf16* __restrict__ xb) {
    int i = (blockIdx.x * 256 + threadIdx.x) * 8;
    f32x4 a = *reinterpret_cast<const f32x4*>(x + i);
    f32x4 b = *reinterpret_cast<const f32x4*>(x + i + 4);
    union { __bf16 h[8]; u32x4 q; } o;
    o.h[0] = f2bf(a.x); o.h[1] = f2bf(a.y); o.h[2] = f2bf(a.z); o.h[3] = f2bf(a.w);
    o.h[4] = f2bf(b.x); o.h[5] = f2bf(b.y); o.h[6] = f2bf(b.z); o.h[7] = f2bf(b.w);
    *reinterpret_cast<u32x4*>(xb + i) = o.q;
}

// W[k][n] fp32 -> WT[n][k] bf16 (tiled transpose through LDS)
__global__ __launch_bounds__(256) void wt_kernel(
    const float* __restrict__ W0, const float* __restrict__ W1,
    const float* __restrict__ W2, const float* __restrict__ W3,
    __bf16* __restrict__ T0, __bf16* __restrict__ T1,
    __bf16* __restrict__ T2, __bf16* __restrict__ T3) {
    const float* W = (blockIdx.z == 0) ? W0 : (blockIdx.z == 1) ? W1
                   : (blockIdx.z == 2) ? W2 : W3;
    __bf16* T = (blockIdx.z == 0) ? T0 : (blockIdx.z == 1) ? T1
              : (blockIdx.z == 2) ? T2 : T3;
    __shared__ float tile[32][33];
    int n0 = blockIdx.x * 32, k0 = blockIdx.y * 32;
#pragma unroll
    for (int r = 0; r < 4; r++) {
        int kk = threadIdx.y + r * 8;
        tile[kk][threadIdx.x] = W[(size_t)(k0 + kk) * kD + n0 + threadIdx.x];
    }
    __syncthreads();
#pragma unroll
    for (int r = 0; r < 4; r++) {
        int nn = threadIdx.y + r * 8;
        T[(size_t)(n0 + nn) * kD + k0 + threadIdx.x] = f2bf(tile[threadIdx.x][nn]);
    }
}

// ---------------- GEMM core: wave computes 32x64 tile of C = A @ W^T^T ----------------

__device__ __forceinline__ void gemm_tile(const __bf16* __restrict__ A,
                                          const __bf16* __restrict__ WT,
                                          int m0, int n0, int lane, v8f acc[2][4]) {
    for (int k = 0; k < kD; k += 32) {
        v16bf a0 = load_afrag(A, kD, m0,      k, lane);
        v16bf a1 = load_afrag(A, kD, m0 + 16, k, lane);
        v16bf b0 = load_bfrag(WT, kD, n0,      k, lane);
        v16bf b1 = load_bfrag(WT, kD, n0 + 16, k, lane);
        v16bf b2 = load_bfrag(WT, kD, n0 + 32, k, lane);
        v16bf b3 = load_bfrag(WT, kD, n0 + 48, k, lane);
        acc[0][0] = wmma_bf16(a0, b0, acc[0][0]);
        acc[0][1] = wmma_bf16(a0, b1, acc[0][1]);
        acc[0][2] = wmma_bf16(a0, b2, acc[0][2]);
        acc[0][3] = wmma_bf16(a0, b3, acc[0][3]);
        acc[1][0] = wmma_bf16(a1, b0, acc[1][0]);
        acc[1][1] = wmma_bf16(a1, b1, acc[1][1]);
        acc[1][2] = wmma_bf16(a1, b2, acc[1][2]);
        acc[1][3] = wmma_bf16(a1, b3, acc[1][3]);
    }
}

// QKV projection: grid.z selects Q/K/V. Q,K -> [B,H,S,DH]; V -> [B,H,DH,S] (transposed).
__global__ __launch_bounds__(256) void qkv_kernel(
    const __bf16* __restrict__ Xb,
    const __bf16* __restrict__ WTq, const __bf16* __restrict__ WTk,
    const __bf16* __restrict__ WTv,
    const float* __restrict__ bq, const float* __restrict__ bk,
    const float* __restrict__ bv,
    __bf16* __restrict__ Qb, __bf16* __restrict__ Kb, __bf16* __restrict__ Vtb) {
    int z = blockIdx.z;
    const __bf16* WT  = (z == 0) ? WTq : (z == 1) ? WTk : WTv;
    const float* bias = (z == 0) ? bq  : (z == 1) ? bk  : bv;
    int lane = threadIdx.x & 31, wid = threadIdx.x >> 5;
    int m0 = blockIdx.x * 64  + (wid >> 2) * 32;
    int n0 = blockIdx.y * 256 + (wid & 3) * 64;
    v8f acc[2][4];
#pragma unroll
    for (int i = 0; i < 2; i++)
#pragma unroll
        for (int j = 0; j < 4; j++)
            acc[i][j] = v8f{0.f, 0.f, 0.f, 0.f, 0.f, 0.f, 0.f, 0.f};
    gemm_tile(Xb, WT, m0, n0, lane, acc);

    int col = lane & 15, rowB = (lane >> 4) * 8;
#pragma unroll
    for (int i = 0; i < 2; i++) {
#pragma unroll
        for (int j = 0; j < 4; j++) {
            int n = n0 + j * 16 + col;
            float bb = bias[n];
            int h = n >> 6, dh = n & 63;
#pragma unroll
            for (int e = 0; e < 8; e++) {
                int tok = m0 + i * 16 + rowB + e;
                int b = tok >> 11, s = tok & (kS - 1);
                __bf16 v = f2bf(acc[i][j][e] + bb);
                if (z == 2)
                    Vtb[((size_t)(b * kH + h) * kDH + dh) * kS + s] = v;
                else if (z == 0)
                    Qb[((size_t)(b * kH + h) * kS + s) * kDH + dh] = v;
                else
                    Kb[((size_t)(b * kH + h) * kS + s) * kDH + dh] = v;
            }
        }
    }
}

// ---------------- flash attention: 4 waves/block, wave = 16-query stripe ----------------

__global__ __launch_bounds__(128) void attn_kernel(
    const __bf16* __restrict__ Qb, const __bf16* __restrict__ Kb,
    const __bf16* __restrict__ Vtb, __bf16* __restrict__ Ctx) {
    __shared__ __align__(16) __bf16 Plds[4][16][64]; // wave-private P tiles
    int lane = threadIdx.x & 31, wid = threadIdx.x >> 5;
    int bh = blockIdx.y;
    int qtile = blockIdx.x * 64;
    int q0 = qtile + wid * 16;
    const __bf16* Qp = Qb  + (size_t)bh * kS  * kDH;
    const __bf16* Kp = Kb  + (size_t)bh * kS  * kDH;
    const __bf16* Vp = Vtb + (size_t)bh * kDH * kS;
    int col = lane & 15, rowB = (lane >> 4) * 8;

    v16bf qf0 = load_afrag(Qp, kDH, q0, 0, lane);
    v16bf qf1 = load_afrag(Qp, kDH, q0, 32, lane);

    float mrun[8], lrun[8];
    v8f ctx[4];
#pragma unroll
    for (int e = 0; e < 8; e++) { mrun[e] = -3.0e38f; lrun[e] = 0.f; }
#pragma unroll
    for (int d = 0; d < 4; d++) ctx[d] = v8f{0.f, 0.f, 0.f, 0.f, 0.f, 0.f, 0.f, 0.f};

    const float scale = 0.125f; // 1/sqrt(64)

    for (int kt = 0; kt <= qtile; kt += 64) {
        // scores: 16 queries x 64 keys, K-frags straight from global (K is B^T layout)
        v8f sc[4];
#pragma unroll
        for (int n = 0; n < 4; n++) {
            v8f s = v8f{0.f, 0.f, 0.f, 0.f, 0.f, 0.f, 0.f, 0.f};
            s = wmma_bf16(qf0, load_bfrag(Kp, kDH, kt + n * 16, 0, lane), s);
            s = wmma_bf16(qf1, load_bfrag(Kp, kDH, kt + n * 16, 32, lane), s);
            sc[n] = s;
        }
        // scale + causal mask + row max (rows live across 16-lane halves)
        float mt[8];
#pragma unroll
        for (int e = 0; e < 8; e++) {
            int qg = q0 + rowB + e;
            float mx = -3.0e38f;
#pragma unroll
            for (int n = 0; n < 4; n++) {
                float v = sc[n][e] * scale;
                if (kt + n * 16 + col > qg) v = -3.0e38f;
                sc[n][e] = v;
                mx = fmaxf(mx, v);
            }
            mx = fmaxf(mx, __shfl_xor(mx, 1));
            mx = fmaxf(mx, __shfl_xor(mx, 2));
            mx = fmaxf(mx, __shfl_xor(mx, 4));
            mx = fmaxf(mx, __shfl_xor(mx, 8));
            mt[e] = mx;
        }
        // online softmax update
#pragma unroll
        for (int e = 0; e < 8; e++) {
            float mnew  = fmaxf(mrun[e], mt[e]);
            float alpha = __expf(mrun[e] - mnew);
            mrun[e] = mnew;
            float rs = 0.f;
#pragma unroll
            for (int n = 0; n < 4; n++) {
                float p = __expf(sc[n][e] - mnew);
                sc[n][e] = p;
                rs += p;
            }
            rs += __shfl_xor(rs, 1);
            rs += __shfl_xor(rs, 2);
            rs += __shfl_xor(rs, 4);
            rs += __shfl_xor(rs, 8);
            lrun[e] = alpha * lrun[e] + rs;
#pragma unroll
            for (int d = 0; d < 4; d++) ctx[d][e] *= alpha;
        }
        // transpose P (C layout -> A layout) through wave-private LDS
#pragma unroll
        for (int n = 0; n < 4; n++)
#pragma unroll
            for (int e = 0; e < 8; e++)
                Plds[wid][rowB + e][n * 16 + col] = f2bf(sc[n][e]);
        const __bf16* pl = &Plds[wid][0][0];
        v16bf pf0 = load_afrag(pl, 64, 0, 0, lane);  // LDS ops in-order per wave
        v16bf pf1 = load_afrag(pl, 64, 0, 32, lane);
        // ctx += P @ V ; V-frags straight from global (Vt is B^T layout)
#pragma unroll
        for (int d = 0; d < 4; d++) {
            ctx[d] = wmma_bf16(pf0, load_bfrag(Vp, kS, d * 16, kt, lane), ctx[d]);
            ctx[d] = wmma_bf16(pf1, load_bfrag(Vp, kS, d * 16, kt + 32, lane), ctx[d]);
        }
    }
    // epilogue: normalize and write ctx as [token][D] bf16 for the O-projection
    int b = bh >> 4, h = bh & 15;
#pragma unroll
    for (int e = 0; e < 8; e++) {
        float rl = 1.0f / lrun[e];
        int qg = q0 + rowB + e;
        size_t base = ((size_t)(b * kS + qg)) * kD + h * kDH;
#pragma unroll
        for (int d = 0; d < 4; d++)
            Ctx[base + d * 16 + col] = f2bf(ctx[d][e] * rl);
    }
}

// ---------------- output projection: fp32 out + bias ----------------

__global__ __launch_bounds__(256) void oproj_kernel(
    const __bf16* __restrict__ Ctx, const __bf16* __restrict__ WTo,
    const float* __restrict__ bo, float* __restrict__ out) {
    int lane = threadIdx.x & 31, wid = threadIdx.x >> 5;
    int m0 = blockIdx.x * 64  + (wid >> 2) * 32;
    int n0 = blockIdx.y * 256 + (wid & 3) * 64;
    v8f acc[2][4];
#pragma unroll
    for (int i = 0; i < 2; i++)
#pragma unroll
        for (int j = 0; j < 4; j++)
            acc[i][j] = v8f{0.f, 0.f, 0.f, 0.f, 0.f, 0.f, 0.f, 0.f};
    gemm_tile(Ctx, WTo, m0, n0, lane, acc);

    int col = lane & 15, rowB = (lane >> 4) * 8;
#pragma unroll
    for (int i = 0; i < 2; i++) {
#pragma unroll
        for (int j = 0; j < 4; j++) {
            int n = n0 + j * 16 + col;
            float bb = bo[n];
#pragma unroll
            for (int e = 0; e < 8; e++) {
                int tok = m0 + i * 16 + rowB + e;
                out[(size_t)tok * kD + n] = acc[i][j][e] + bb;
            }
        }
    }
}

// ---------------- host launcher ----------------

extern "C" void kernel_launch(void* const* d_in, const int* in_sizes, int n_in,
                              void* d_out, int out_size, void* d_ws, size_t ws_size,
                              hipStream_t stream) {
    (void)in_sizes; (void)n_in; (void)out_size; (void)ws_size;
    const float* hidden = (const float*)d_in[0];
    const float* Wq = (const float*)d_in[1];
    const float* bq = (const float*)d_in[2];
    const float* Wk = (const float*)d_in[3];
    const float* bk = (const float*)d_in[4];
    const float* Wv = (const float*)d_in[5];
    const float* bv = (const float*)d_in[6];
    const float* Wo = (const float*)d_in[7];
    const float* bo = (const float*)d_in[8];

    char* w = (char*)d_ws;
    auto carve = [&](size_t bytes) {
        char* p = w;
        w += (bytes + 255) & ~(size_t)255;
        return p;
    };
    __bf16* Xb  = (__bf16*)carve((size_t)kBS * kD * 2);
    __bf16* WTq = (__bf16*)carve((size_t)kD * kD * 2);
    __bf16* WTk = (__bf16*)carve((size_t)kD * kD * 2);
    __bf16* WTv = (__bf16*)carve((size_t)kD * kD * 2);
    __bf16* WTo = (__bf16*)carve((size_t)kD * kD * 2);
    __bf16* Qb  = (__bf16*)carve((size_t)kBS * kD * 2);
    __bf16* Kb  = (__bf16*)carve((size_t)kBS * kD * 2);
    __bf16* Vtb = (__bf16*)carve((size_t)kBS * kD * 2);
    __bf16* Ctx = (__bf16*)carve((size_t)kBS * kD * 2);

    // 1) convert activations to bf16
    cvt_x_kernel<<<dim3((kBS * kD) / (256 * 8)), dim3(256), 0, stream>>>(hidden, Xb);
    // 2) transpose + convert weights (W[k][n] -> WT[n][k] bf16)
    wt_kernel<<<dim3(kD / 32, kD / 32, 4), dim3(32, 8), 0, stream>>>(
        Wq, Wk, Wv, Wo, WTq, WTk, WTv, WTo);
    // 3) fused QKV projections (z = Q/K/V)
    qkv_kernel<<<dim3(kBS / 64, kD / 256, 3), dim3(256), 0, stream>>>(
        Xb, WTq, WTk, WTv, bq, bk, bv, Qb, Kb, Vtb);
    // 4) causal flash attention
    attn_kernel<<<dim3(kS / 64, kB * kH), dim3(128), 0, stream>>>(Qb, Kb, Vtb, Ctx);
    // 5) output projection -> fp32
    oproj_kernel<<<dim3(kBS / 64, kD / 256), dim3(256), 0, stream>>>(
        Ctx, WTo, bo, (float*)d_out);
}
